// GATBlock_76141180223555
// MI455X (gfx1250) — matrix operand order repaired and checked by reading, not compiled, not run on hardware
//
#include <hip/hip_runtime.h>
#include <hip/hip_bf16.h>

// ---------------- problem constants (match reference) ----------------
#define NN   32768      // total nodes
#define FIN  256        // input features
#define HC   256        // H*C output features
#define NH   4          // heads
#define CH   64         // per-head channels
#define BG   64         // graphs
#define NPG  512        // nodes per graph
#define KP   256        // top-k kept per graph

typedef __attribute__((ext_vector_type(16))) __bf16 v16bf;
typedef __attribute__((ext_vector_type(8)))  __bf16 v8bf;
typedef __attribute__((ext_vector_type(8)))  float  v8f;

// ---------------- helpers ----------------
__device__ __forceinline__ void atomicMaxF(float* addr, float val) {
    unsigned int* ua = (unsigned int*)addr;
    unsigned int old = *ua;
    while (__uint_as_float(old) < val) {
        unsigned int assumed = old;
        old = atomicCAS(ua, assumed, __float_as_uint(val));
        if (old == assumed) break;
    }
}

// ---------------- 1) convert x -> bf16 ----------------
__global__ void k_cvt_x(const float* __restrict__ x, __bf16* __restrict__ xb) {
    int i = blockIdx.x * blockDim.x + threadIdx.x;   // N*FIN threads
    xb[i] = (__bf16)x[i];
}

// ---------------- 2) convert W -> bf16, transposed wt[col][f] ----------------
__global__ void k_cvt_w(const float* __restrict__ W, __bf16* __restrict__ wt) {
    int i = blockIdx.x * blockDim.x + threadIdx.x;   // HC*FIN threads
    int col = i / FIN, f = i % FIN;
    wt[(size_t)col * FIN + f] = (__bf16)W[(size_t)f * HC + col];
}

// ---------------- 3) WMMA GEMM: h[N][HC] = xb[N][FIN] * W ----------------
// block = 4 waves; each wave: one A fragment reused across 4 N-tiles
// (4 accumulators, 32 WMMAs); block covers all 16 N-tiles of one M-tile.
__global__ void k_gemm(const __bf16* __restrict__ xb,
                       const __bf16* __restrict__ wt,
                       float* __restrict__ h) {
    int wave = threadIdx.x >> 5;         // 0..3
    int lane = threadIdx.x & 31;
    int mt   = blockIdx.x;               // 0..2047 (M tile)
    int nt0  = wave * 4;                 // this wave: N tiles nt0..nt0+3
    int m    = lane & 15;
    int half = lane >> 4;

    // A: 16x32 bf16 — lane (m, half): K = half*16 + {0..15}
    const __bf16* arow = xb + (size_t)(mt * 16 + m) * FIN;
    // B: 32x16 bf16 — lane (n, half): K = half*16 + {0..15}, contiguous in wt[n][k]
    const __bf16* bbase = wt + (size_t)(nt0 * 16 + m) * FIN + half * 16;

    v8f acc0 = {}, acc1 = {}, acc2 = {}, acc3 = {};
    #pragma unroll
    for (int k0 = 0; k0 < FIN; k0 += 32) {
        v8bf alo = *reinterpret_cast<const v8bf*>(arow + k0 + half * 16);
        v8bf ahi = *reinterpret_cast<const v8bf*>(arow + k0 + half * 16 + 8);
        v16bf a;
        #pragma unroll
        for (int i = 0; i < 8; ++i) { a[i] = alo[i]; a[i + 8] = ahi[i]; }

        v16bf b0 = *reinterpret_cast<const v16bf*>(bbase + k0);
        v16bf b1 = *reinterpret_cast<const v16bf*>(bbase + (size_t)16 * FIN + k0);
        v16bf b2 = *reinterpret_cast<const v16bf*>(bbase + (size_t)32 * FIN + k0);
        v16bf b3 = *reinterpret_cast<const v16bf*>(bbase + (size_t)48 * FIN + k0);

        acc0 = __builtin_amdgcn_wmma_f32_16x16x32_bf16(false, a, false, b0, (short)0, acc0, false, false);
        acc1 = __builtin_amdgcn_wmma_f32_16x16x32_bf16(false, a, false, b1, (short)0, acc1, false, false);
        acc2 = __builtin_amdgcn_wmma_f32_16x16x32_bf16(false, a, false, b2, (short)0, acc2, false, false);
        acc3 = __builtin_amdgcn_wmma_f32_16x16x32_bf16(false, a, false, b3, (short)0, acc3, false, false);
    }
    // C/D layout: VGPR r -> row = r + half*8, col = lane&15 (per N-tile)
    float* hb = h + (size_t)(mt * 16) * HC + nt0 * 16 + (lane & 15);
    #pragma unroll
    for (int r = 0; r < 8; ++r) {
        size_t ro = (size_t)(r + half * 8) * HC;
        hb[ro]      = acc0[r];
        hb[ro + 16] = acc1[r];
        hb[ro + 32] = acc2[r];
        hb[ro + 48] = acc3[r];
    }
}

// ---------------- 4) attention values a_src/a_dst [N][NH] ----------------
__global__ void k_att(const float* __restrict__ h,
                      const float* __restrict__ att_src,
                      const float* __restrict__ att_dst,
                      float* __restrict__ a_src,
                      float* __restrict__ a_dst) {
    int gid = blockIdx.x * blockDim.x + threadIdx.x;   // N*NH threads
    int n = gid >> 2, hh = gid & 3;
    const float* hr = h + (size_t)n * HC + hh * CH;
    const float* as = att_src + hh * CH;
    const float* ad = att_dst + hh * CH;
    float s0 = 0.f, s1 = 0.f;
    #pragma unroll 8
    for (int c = 0; c < CH; ++c) { float v = hr[c]; s0 += v * as[c]; s1 += v * ad[c]; }
    a_src[gid] = s0;
    a_dst[gid] = s1;
}

// ---------------- 5) init emax / den ----------------
__global__ void k_init(float* __restrict__ emax, float* __restrict__ den) {
    int i = blockIdx.x * blockDim.x + threadIdx.x;   // N*NH threads
    emax[i] = -3.0e38f;
    den[i]  = 0.f;
}

// ---------------- 6) edge pass: segment max ----------------
__global__ void k_edge_max(const int* __restrict__ ei, int E, int ET,
                           const float* __restrict__ a_src,
                           const float* __restrict__ a_dst,
                           float* __restrict__ emax) {
    int e = blockIdx.x * blockDim.x + threadIdx.x;
    if (e >= ET) return;
    int s, d;
    if (e < E) { s = ei[e]; d = ei[E + e]; } else { s = d = e - E; }
    #pragma unroll
    for (int hh = 0; hh < NH; ++hh) {
        float v = a_src[s * NH + hh] + a_dst[d * NH + hh];
        v = v > 0.f ? v : 0.2f * v;                 // leaky_relu 0.2
        atomicMaxF(&emax[d * NH + hh], v);
    }
}

// ---------------- 7) edge pass: exp + segment sum ----------------
__global__ void k_edge_exp(const int* __restrict__ ei, int E, int ET,
                           const float* __restrict__ a_src,
                           const float* __restrict__ a_dst,
                           const float* __restrict__ emax,
                           float* __restrict__ den,
                           float* __restrict__ ex) {
    int e = blockIdx.x * blockDim.x + threadIdx.x;
    if (e >= ET) return;
    int s, d;
    if (e < E) { s = ei[e]; d = ei[E + e]; } else { s = d = e - E; }
    #pragma unroll
    for (int hh = 0; hh < NH; ++hh) {
        float v = a_src[s * NH + hh] + a_dst[d * NH + hh];
        v = v > 0.f ? v : 0.2f * v;
        float xv = expf(v - emax[d * NH + hh]);
        ex[(size_t)e * NH + hh] = xv;
        unsafeAtomicAdd(&den[d * NH + hh], xv);
    }
}

// ---------------- 8) edge aggregate: agg[dst] += alpha * h[src] ----------------
// one wave per edge; lane handles 8 consecutive channels (all in one head)
__global__ void k_edge_agg(const int* __restrict__ ei, int E, int ET,
                           const float* __restrict__ ex,
                           const float* __restrict__ den,
                           const float* __restrict__ h,
                           float* __restrict__ agg) {
    int widx = (blockIdx.x * blockDim.x + threadIdx.x) >> 5;  // edge id
    int lane = threadIdx.x & 31;
    if (widx >= ET) return;
    int s, d;
    if (widx < E) { s = ei[widx]; d = ei[E + widx]; } else { s = d = widx - E; }
    int hh = lane >> 3;                                       // head of lane's 8 channels
    float alpha = ex[(size_t)widx * NH + hh] / (den[d * NH + hh] + 1e-16f);
    const float4* hp = reinterpret_cast<const float4*>(h + (size_t)s * HC + lane * 8);
    float4 v0 = hp[0], v1 = hp[1];
    float* ob = agg + (size_t)d * HC + lane * 8;
    unsafeAtomicAdd(ob + 0, alpha * v0.x);
    unsafeAtomicAdd(ob + 1, alpha * v0.y);
    unsafeAtomicAdd(ob + 2, alpha * v0.z);
    unsafeAtomicAdd(ob + 3, alpha * v0.w);
    unsafeAtomicAdd(ob + 4, alpha * v1.x);
    unsafeAtomicAdd(ob + 5, alpha * v1.y);
    unsafeAtomicAdd(ob + 6, alpha * v1.z);
    unsafeAtomicAdd(ob + 7, alpha * v1.w);
}

// ---------------- 9) bias + relu (in place) + per-channel stats ----------------
// grid 256 blocks x 128 rows; thread = channel
__global__ void k_relu_stats(float* __restrict__ agg,
                             const float* __restrict__ bias,
                             float* __restrict__ chsum,
                             float* __restrict__ chsum2) {
    int c = threadIdx.x;
    int r0 = blockIdx.x * 128;
    float b = bias[c];
    float s = 0.f, s2 = 0.f;
    for (int r = 0; r < 128; ++r) {
        size_t idx = (size_t)(r0 + r) * HC + c;
        float v = agg[idx] + b;
        v = v > 0.f ? v : 0.f;
        agg[idx] = v;
        s += v; s2 += v * v;
    }
    unsafeAtomicAdd(&chsum[c], s);
    unsafeAtomicAdd(&chsum2[c], s2);
}

// ---------------- 10) BN params + normalized pool weight ----------------
__global__ void k_bn_params(const float* __restrict__ chsum,
                            const float* __restrict__ chsum2,
                            const float* __restrict__ gamma,
                            const float* __restrict__ beta,
                            const float* __restrict__ pw,
                            float* __restrict__ scale,
                            float* __restrict__ shift,
                            float* __restrict__ wsc) {
    __shared__ float red[HC];
    int c = threadIdx.x;
    float invN = 1.0f / (float)NN;
    float mean = chsum[c] * invN;
    float var  = chsum2[c] * invN - mean * mean;
    float sc = gamma[c] * rsqrtf(var + 1e-5f);
    scale[c] = sc;
    shift[c] = beta[c] - mean * sc;
    float w = pw[c];
    red[c] = w * w;
    __syncthreads();
    for (int off = HC / 2; off > 0; off >>= 1) {
        if (c < off) red[c] += red[c + off];
        __syncthreads();
    }
    float invn = rsqrtf(red[0]);
    wsc[c] = w * invn;
}

// ---------------- 11) apply BN (in place) + score per node ----------------
__global__ void k_bn_score(float* __restrict__ agg,
                           const float* __restrict__ scale,
                           const float* __restrict__ shift,
                           const float* __restrict__ wsc,
                           float* __restrict__ score) {
    __shared__ float red[HC];
    int row = blockIdx.x, c = threadIdx.x;
    size_t idx = (size_t)row * HC + c;
    float fb = agg[idx] * scale[c] + shift[c];
    agg[idx] = fb;
    red[c] = fb * wsc[c];
    __syncthreads();
    for (int off = HC / 2; off > 0; off >>= 1) {
        if (c < off) red[c] += red[c + off];
        __syncthreads();
    }
    if (c == 0) score[row] = red[0];
}

// ---------------- 12) per-graph bitonic top-k (descending, stable by index) ----
__global__ void k_topk(const float* __restrict__ score,
                       float* __restrict__ tops_s,
                       int* __restrict__ tops_i) {
    __shared__ float ss[NPG];
    __shared__ int   si[NPG];
    int b = blockIdx.x, t = threadIdx.x;
    ss[t] = score[b * NPG + t];
    si[t] = t;
    __syncthreads();
    for (int k = 2; k <= NPG; k <<= 1) {
        for (int j = k >> 1; j > 0; j >>= 1) {
            int ixj = t ^ j;
            if (ixj > t) {
                float s1 = ss[t], s2 = ss[ixj];
                int   i1 = si[t], i2 = si[ixj];
                // desired order: score desc, index asc on ties
                bool firstBefore = (s1 > s2) || (s1 == s2 && i1 < i2);
                bool dirDesc = ((t & k) == 0);
                bool doswap = dirDesc ? !firstBefore : firstBefore;
                if (doswap) { ss[t] = s2; ss[ixj] = s1; si[t] = i2; si[ixj] = i1; }
            }
            __syncthreads();
        }
    }
    if (t < KP) {
        tops_s[b * KP + t] = ss[t];
        tops_i[b * KP + t] = si[t];
    }
}

// ---------------- 13) gather + tanh gate ----------------
__global__ void k_gather(const float* __restrict__ agg,
                         const float* __restrict__ tops_s,
                         const int* __restrict__ tops_i,
                         float* __restrict__ out) {
    int row = blockIdx.x, c = threadIdx.x;   // row in [0, BG*KP)
    int b = row / KP;
    int node = b * NPG + tops_i[row];
    float g = tanhf(tops_s[row]);
    out[(size_t)row * HC + c] = agg[(size_t)node * HC + c] * g;
}

// ---------------- launcher ----------------
extern "C" void kernel_launch(void* const* d_in, const int* in_sizes, int n_in,
                              void* d_out, int out_size, void* d_ws, size_t ws_size,
                              hipStream_t stream) {
    const float* x        = (const float*)d_in[0];
    const int*   ei       = (const int*)d_in[1];
    const float* W        = (const float*)d_in[2];
    const float* att_src  = (const float*)d_in[3];
    const float* att_dst  = (const float*)d_in[4];
    const float* bias     = (const float*)d_in[5];
    const float* bn_gamma = (const float*)d_in[6];
    const float* bn_beta  = (const float*)d_in[7];
    const float* pw       = (const float*)d_in[8];
    float* out = (float*)d_out;

    const int E  = in_sizes[1] / 2;     // 524288
    const int ET = E + NN;              // + self loops

    // ---- workspace carve-up (256B aligned) ----
    char* ws = (char*)d_ws;
    size_t off = 0;
    auto carve = [&](size_t bytes) {
        size_t o = off;
        off = (off + bytes + 255) & ~(size_t)255;
        return o;
    };
    // region A: xb (bf16, used only through GEMM) overlaid with ex (f32, used after)
    size_t xb_bytes = (size_t)NN * FIN * sizeof(__bf16);          // 16 MB
    size_t ex_bytes = (size_t)ET * NH * sizeof(float);            // 8.9 MB
    size_t regA = carve(xb_bytes > ex_bytes ? xb_bytes : ex_bytes);
    __bf16* xb = (__bf16*)(ws + regA);
    float*  ex = (float*)(ws + regA);
    __bf16* wt    = (__bf16*)(ws + carve((size_t)HC * FIN * sizeof(__bf16)));
    float*  h     = (float*)(ws + carve((size_t)NN * HC * sizeof(float)));
    float*  agg   = (float*)(ws + carve((size_t)NN * HC * sizeof(float)));
    float*  a_src = (float*)(ws + carve((size_t)NN * NH * sizeof(float)));
    float*  a_dst = (float*)(ws + carve((size_t)NN * NH * sizeof(float)));
    float*  emax  = (float*)(ws + carve((size_t)NN * NH * sizeof(float)));
    float*  den   = (float*)(ws + carve((size_t)NN * NH * sizeof(float)));
    float*  chsum = (float*)(ws + carve(2 * HC * sizeof(float)));   // sum + sumsq
    float*  chsum2 = chsum + HC;
    float*  scale = (float*)(ws + carve(HC * sizeof(float)));
    float*  shiftv= (float*)(ws + carve(HC * sizeof(float)));
    float*  wsc   = (float*)(ws + carve(HC * sizeof(float)));
    float*  score = (float*)(ws + carve((size_t)NN * sizeof(float)));
    float*  tops_s= (float*)(ws + carve((size_t)BG * KP * sizeof(float)));
    int*    tops_i= (int*)(ws + carve((size_t)BG * KP * sizeof(int)));
    (void)ws_size; (void)n_in; (void)out_size;

    // 1-2) bf16 conversions
    k_cvt_x<<<(NN * FIN) / 256, 256, 0, stream>>>(x, xb);
    k_cvt_w<<<(HC * FIN) / 256, 256, 0, stream>>>(W, wt);

    // 3) WMMA GEMM (one block per 16-row M tile; 4 waves x 4 N-tiles)
    k_gemm<<<NN / 16, 128, 0, stream>>>(xb, wt, h);

    // 4) attention logits
    k_att<<<(NN * NH) / 256, 256, 0, stream>>>(h, att_src, att_dst, a_src, a_dst);

    // 5) init reductions
    k_init<<<(NN * NH) / 256, 256, 0, stream>>>(emax, den);
    hipMemsetAsync(agg, 0, (size_t)NN * HC * sizeof(float), stream);
    hipMemsetAsync(chsum, 0, 2 * HC * sizeof(float), stream);

    // 6-8) edge passes
    int egrid = (ET + 255) / 256;
    k_edge_max<<<egrid, 256, 0, stream>>>(ei, E, ET, a_src, a_dst, emax);
    k_edge_exp<<<egrid, 256, 0, stream>>>(ei, E, ET, a_src, a_dst, emax, den, ex);
    k_edge_agg<<<(ET + 7) / 8, 256, 0, stream>>>(ei, E, ET, ex, den, h, agg);

    // 9-11) bias+relu, BN stats, BN apply + score
    k_relu_stats<<<NN / 128, 256, 0, stream>>>(agg, bias, chsum, chsum2);
    k_bn_params<<<1, HC, 0, stream>>>(chsum, chsum2, bn_gamma, bn_beta, pw,
                                      scale, shiftv, wsc);
    k_bn_score<<<NN, HC, 0, stream>>>(agg, scale, shiftv, wsc, score);

    // 12-13) top-k + gather
    k_topk<<<BG, NPG, 0, stream>>>(score, tops_s, tops_i);
    k_gather<<<BG * KP, HC, 0, stream>>>(agg, tops_s, tops_i, out);
}
